// SparseConv3d_4415226380608
// MI455X (gfx1250) — matrix operand order repaired and checked by reading, not compiled, not run on hardware
//
#include <hip/hip_runtime.h>

typedef _Float16 v16h __attribute__((ext_vector_type(16)));
typedef _Float16 v8h  __attribute__((ext_vector_type(8)));
typedef _Float16 v4h  __attribute__((ext_vector_type(4)));
typedef float    v8f  __attribute__((ext_vector_type(8)));
typedef float    v4f  __attribute__((ext_vector_type(4)));

#define N_PTS 262144
#define KOFF  27
#define CIN   64
#define COUT  64

// ---------------------------------------------------------------------------
// Kernel 1: convert features f32 -> f16, append one zero pad row at index N.
// ---------------------------------------------------------------------------
__global__ __launch_bounds__(256) void pack_features(
    const float* __restrict__ f, _Float16* __restrict__ out)
{
    int t = blockIdx.x * blockDim.x + threadIdx.x;      // chunk of 4 elements
    const int total = (N_PTS + 1) * CIN / 4;
    if (t >= total) return;
    int base = t * 4;
    v4f v = {0.f, 0.f, 0.f, 0.f};
    if (base < N_PTS * CIN) v = *(const v4f*)(f + base);
    v4h h;
    h.x = (_Float16)v.x; h.y = (_Float16)v.y;
    h.z = (_Float16)v.z; h.w = (_Float16)v.w;
    *(v4h*)(out + base) = h;
}

// ---------------------------------------------------------------------------
// Kernel 2: pre-swizzle weights into WMMA B-fragment layout.
// Fragment = 32(K) x 16(N) f16 for v_wmma_f32_16x16x32_f16:
//   lane L holds column n = L&15; K = (L>>4)*16 + e (contiguous order chosen here, same as A).
// Layout in ws: bpack[k][s][nt][lane][16 halves], frag stride = 512 halves,
// per-k block = 4096 halves (8 KB).
// ---------------------------------------------------------------------------
__global__ __launch_bounds__(256) void pack_weights(
    const float* __restrict__ w, _Float16* __restrict__ bpack)
{
    int k    = blockIdx.x;          // 0..26
    int t    = threadIdx.x;         // 0..255 -> 8 frags x 32 lanes
    int frag = t >> 5;              // s*4 + nt
    int lane = t & 31;
    int s    = frag >> 2;
    int nt   = frag & 3;
    int n    = nt * 16 + (lane & 15);
    int kbase = s * 32 + (lane >> 4) * 16;

    _Float16* dst = bpack + ((size_t)k * 8 + frag) * 512 + (size_t)lane * 16;
    const float* src = w + (size_t)k * CIN * COUT;
#pragma unroll
    for (int e = 0; e < 16; ++e)
        dst[e] = (_Float16)src[(size_t)(kbase + e) * COUT + n];
}

// ---------------------------------------------------------------------------
// Kernel 3: main gathered GEMM.
// 8 waves/block, M=32 rows per wave (256 rows per block), C_out=64 wide.
// B fragments double-buffered through LDS (2 x 8 KB); one barrier per k.
// ---------------------------------------------------------------------------
__global__ __launch_bounds__(256) void conv_wmma(
    const _Float16* __restrict__ featH,   // [(N+1) x 64] f16, row N = zeros
    const int*      __restrict__ nbr,     // [K x N]
    const _Float16* __restrict__ bpack,   // swizzled weights, 8 KB per k
    float*          __restrict__ out)     // [N x 64] f32
{
    __shared__ _Float16 Bsh[2][4096];     // 2 x 8 KB double buffer

    const int t       = threadIdx.x;
    const int wave    = t >> 5;
    const int lane    = t & 31;
    const int halfSel = lane >> 4;        // 0 | 1
    const int laneM   = lane & 15;
    const int rowBase = blockIdx.x * 256 + wave * 32;
    const int rowA0   = rowBase + laneM;        // m-tile 0 gather row
    const int rowA1   = rowBase + 16 + laneM;   // m-tile 1 gather row

    v8f c[8];                              // [mt][nt] accumulators
#pragma unroll
    for (int i = 0; i < 8; ++i) c[i] = (v8f){};

    // stage k = 0 into buffer 0 (one 32 B chunk per thread)
    {
        v16h tmp = *(const v16h*)(bpack + (size_t)t * 16);
        *(v16h*)(&Bsh[0][t * 16]) = tmp;
    }
    __syncthreads();

    for (int k = 0; k < KOFF; ++k) {
        const int cur = k & 1;

        // issue prefetch of next k's B block (global -> regs) early
        v16h nxt;
        const bool have_next = (k + 1 < KOFF);
        if (have_next)
            nxt = *(const v16h*)(bpack + (size_t)(k + 1) * 4096 + (size_t)t * 16);

        const int idx0 = nbr[(size_t)k * N_PTS + rowA0];
        const int idx1 = nbr[(size_t)k * N_PTS + rowA1];
        const _Float16* rp0 = featH + (size_t)idx0 * CIN;
        const _Float16* rp1 = featH + (size_t)idx1 * CIN;

#pragma unroll
        for (int s = 0; s < 2; ++s) {
            // A fragments (16x32 f16): lane holds its row's K chunk
            // 16*half .. 16*half+15 contiguous (matches B packing).
            const int abase = s * 32 + halfSel * 16;
            v8h lo0 = *(const v8h*)(rp0 + abase);
            v8h hi0 = *(const v8h*)(rp0 + abase + 8);
            v8h lo1 = *(const v8h*)(rp1 + abase);
            v8h hi1 = *(const v8h*)(rp1 + abase + 8);
            v16h a0, a1;
#pragma unroll
            for (int i = 0; i < 8; ++i) {
                a0[i] = lo0[i]; a0[8 + i] = hi0[i];
                a1[i] = lo1[i]; a1[8 + i] = hi1[i];
            }

            const _Float16* bs = &Bsh[cur][s * 2048 + lane * 16];
            v16h b0 = *(const v16h*)(bs);
            v16h b1 = *(const v16h*)(bs + 512);
            v16h b2 = *(const v16h*)(bs + 1024);
            v16h b3 = *(const v16h*)(bs + 1536);

            c[0] = __builtin_amdgcn_wmma_f32_16x16x32_f16(false, a0, false, b0,
                                                          (short)0, c[0], false, false);
            c[1] = __builtin_amdgcn_wmma_f32_16x16x32_f16(false, a0, false, b1,
                                                          (short)0, c[1], false, false);
            c[2] = __builtin_amdgcn_wmma_f32_16x16x32_f16(false, a0, false, b2,
                                                          (short)0, c[2], false, false);
            c[3] = __builtin_amdgcn_wmma_f32_16x16x32_f16(false, a0, false, b3,
                                                          (short)0, c[3], false, false);
            c[4] = __builtin_amdgcn_wmma_f32_16x16x32_f16(false, a1, false, b0,
                                                          (short)0, c[4], false, false);
            c[5] = __builtin_amdgcn_wmma_f32_16x16x32_f16(false, a1, false, b1,
                                                          (short)0, c[5], false, false);
            c[6] = __builtin_amdgcn_wmma_f32_16x16x32_f16(false, a1, false, b2,
                                                          (short)0, c[6], false, false);
            c[7] = __builtin_amdgcn_wmma_f32_16x16x32_f16(false, a1, false, b3,
                                                          (short)0, c[7], false, false);
        }

        // write next k's B into the other LDS buffer, then sync once
        if (have_next)
            *(v16h*)(&Bsh[cur ^ 1][t * 16]) = nxt;
        __syncthreads();
    }

    // C/D layout: lanes 0-15: VGPR r -> M=r, N=lane; lanes 16-31: M=r+8.
#pragma unroll
    for (int mt = 0; mt < 2; ++mt) {
        const int row0 = rowBase + mt * 16 + halfSel * 8;
#pragma unroll
        for (int r = 0; r < 8; ++r) {
            float* o = out + (size_t)(row0 + r) * COUT + laneM;
            o[0]  = c[mt * 4 + 0][r];
            o[16] = c[mt * 4 + 1][r];
            o[32] = c[mt * 4 + 2][r];
            o[48] = c[mt * 4 + 3][r];
        }
    }
}

// ---------------------------------------------------------------------------
// Kernel 4a: per-1024-row partial sums / sumsq per channel (coalesced).
// ---------------------------------------------------------------------------
__global__ __launch_bounds__(256) void stats_partial(
    const float* __restrict__ out, float* __restrict__ partial)
{
    __shared__ float shs[256];
    __shared__ float shq[256];
    const int t  = threadIdx.x;
    const int c  = t & 63;
    const int rr = t >> 6;                      // 0..3
    const size_t base = (size_t)blockIdx.x * 1024;

    float s = 0.f, sq = 0.f;
    for (int j = 0; j < 256; ++j) {
        float v = out[(base + rr + 4 * (size_t)j) * COUT + c];
        s += v; sq += v * v;
    }
    shs[t] = s; shq[t] = sq;
    __syncthreads();
    if (rr == 0) {
        s  = shs[c] + shs[c + 64] + shs[c + 128] + shs[c + 192];
        sq = shq[c] + shq[c + 64] + shq[c + 128] + shq[c + 192];
        partial[(size_t)blockIdx.x * 128 + c]      = s;
        partial[(size_t)blockIdx.x * 128 + 64 + c] = sq;
    }
}

// ---------------------------------------------------------------------------
// Kernel 4b: finalize mean/var -> per-channel scale/shift.
// ---------------------------------------------------------------------------
__global__ void stats_final(
    const float* __restrict__ partial, const float* __restrict__ gamma,
    const float* __restrict__ beta, float* __restrict__ ss)
{
    int c = threadIdx.x;
    if (c >= 64) return;
    float s = 0.f, sq = 0.f;
    for (int b = 0; b < 256; ++b) {
        s  += partial[(size_t)b * 128 + c];
        sq += partial[(size_t)b * 128 + 64 + c];
    }
    const float invN = 1.f / (float)N_PTS;
    float mean  = s * invN;
    float var   = sq * invN - mean * mean;
    float scale = gamma[c] * rsqrtf(var + 1e-5f);
    ss[c]      = scale;
    ss[64 + c] = beta[c] - mean * scale;
}

// ---------------------------------------------------------------------------
// Kernel 5: in-place BN + ReLU, float4 vectorized.
// ---------------------------------------------------------------------------
__global__ __launch_bounds__(256) void bn_relu(
    float* __restrict__ out, const float* __restrict__ ss)
{
    size_t t    = (size_t)blockIdx.x * blockDim.x + threadIdx.x;
    size_t base = t * 4;
    int c = (int)(base & 63);                  // multiple of 4
    v4f v  = *(v4f*)(out + base);
    v4f sc = *(const v4f*)(ss + c);
    v4f sh = *(const v4f*)(ss + 64 + c);
    v.x = fmaxf(v.x * sc.x + sh.x, 0.f);
    v.y = fmaxf(v.y * sc.y + sh.y, 0.f);
    v.z = fmaxf(v.z * sc.z + sh.z, 0.f);
    v.w = fmaxf(v.w * sc.w + sh.w, 0.f);
    *(v4f*)(out + base) = v;
}

// ---------------------------------------------------------------------------
extern "C" void kernel_launch(void* const* d_in, const int* in_sizes, int n_in,
                              void* d_out, int out_size, void* d_ws, size_t ws_size,
                              hipStream_t stream)
{
    (void)in_sizes; (void)n_in; (void)out_size; (void)ws_size;

    const float* features = (const float*)d_in[0];   // [N, 64] f32
    const int*   nbr      = (const int*)  d_in[1];   // [27, N] i32
    const float* weight   = (const float*)d_in[2];   // [27, 64, 64] f32
    const float* gamma    = (const float*)d_in[3];   // [64]
    const float* beta     = (const float*)d_in[4];   // [64]
    float*       out      = (float*)d_out;           // [N, 64] f32

    char* ws = (char*)d_ws;
    // workspace layout (256B aligned sections)
    size_t featBytes  = (size_t)(N_PTS + 1) * CIN * sizeof(_Float16); // 33,554,560
    size_t bpackOff   = (featBytes + 255) & ~(size_t)255;
    size_t bpackBytes = (size_t)KOFF * 8 * 512 * sizeof(_Float16);   // 221,184
    size_t partOff    = (bpackOff + bpackBytes + 255) & ~(size_t)255;
    size_t partBytes  = (size_t)256 * 128 * sizeof(float);           // 131,072
    size_t ssOff      = (partOff + partBytes + 255) & ~(size_t)255;

    _Float16* featH  = (_Float16*)(ws);
    _Float16* bpack  = (_Float16*)(ws + bpackOff);
    float*    part   = (float*)(ws + partOff);
    float*    ss     = (float*)(ws + ssOff);

    const int pfChunks = (N_PTS + 1) * CIN / 4;
    pack_features<<<(pfChunks + 255) / 256, 256, 0, stream>>>(features, featH);
    pack_weights<<<KOFF, 256, 0, stream>>>(weight, bpack);
    conv_wmma<<<N_PTS / 256, 256, 0, stream>>>(featH, nbr, bpack, out);
    stats_partial<<<N_PTS / 1024, 256, 0, stream>>>(out, part);
    stats_final<<<1, 64, 0, stream>>>(part, gamma, beta, ss);
    bn_relu<<<(N_PTS * COUT / 4) / 256, 256, 0, stream>>>(out, ss);
}
